// HRM_4x4_18760417149191
// MI455X (gfx1250) — matrix operand order repaired and checked by reading, not compiled, not run on hardware
//
#include <hip/hip_runtime.h>
#include <math.h>

typedef _Float16 half_t;
typedef __attribute__((ext_vector_type(16))) _Float16 v16h;
typedef __attribute__((ext_vector_type(8)))  _Float16 v8h;
typedef __attribute__((ext_vector_type(8)))  float    v8f;

#define THREADS      128
#define ROWS_PER_WG  64

// packed-weight offsets (in halfs) within one parameter set
#define WV_H   0
#define OW_H   4096
#define F1_H   8192
#define F2_H   24576
#define SET_H  40960            // 81920 bytes per set, 2 sets

// byte offsets in dynamic LDS
#define BIAS_OFF    163840      // 2 * 576 floats = 4608 B
#define WAVE_OFF    168448
#define WAVE_STRIDE 30720
#define HL_B    0               // 16x64 f32
#define HH_B    4096            // 16x64 f32
#define XIN_B   8192            // 16x64 f32
#define CB_B    12288           // 16x64 f32 (block input / h)
#define OB_B    16384           // 16x64 f32 (matmul out)
#define A64_B   20480           // 16x64 f16 (A staging)
#define A256_B  22528           // 16x256 f16 (hidden / emb staging)
#define LDS_BYTES 291328

struct Params {
  const int*   puzzle;
  const float* emb;   const float* proj_w; const float* proj_b;
  const float* hL0;   const float* hH0;
  const float* cell_w; const float* cell_b; const float* dig_w; const float* dig_b;
  const float* w_inw; const float* w_inb; const float* w_ow;  const float* w_ob;
  const float* w_f1w; const float* w_f1b; const float* w_f2w; const float* w_f2b;
  const float* w_n1;  const float* w_n2;
  const float* p_inw; const float* p_inb; const float* p_ow;  const float* p_ob;
  const float* p_f1w; const float* p_f1b; const float* p_f2w; const float* p_f2b;
  const float* p_n1;  const float* p_n2;
  float* out; int nrows;
};

__device__ __forceinline__ v16h make_v16(v8h lo, v8h hi) {
  return __builtin_shufflevector(lo, hi, 0,1,2,3,4,5,6,7,8,9,10,11,12,13,14,15);
}

// A fragment (16x32 f16): lane%16 = M row; lanes 0-15 hold K {0..7,16..23},
// lanes 16-31 hold K {8..15,24..31} of the 32-wide K block.
__device__ __forceinline__ v16h load_a_frag(const half_t* base, int stride, int kb, int lane) {
  const int M    = lane & 15;
  const int ksel = (lane >> 4) << 3;
  const half_t* p = base + M * stride + kb * 32 + ksel;
  v8h lo = *(const v8h*)p;
  v8h hi = *(const v8h*)(p + 16);
  return make_v16(lo, hi);
}

// B fragment: pre-packed so each lane's 16 halves are contiguous (32 B).
__device__ __forceinline__ v16h load_b_frag(const half_t* tile, int lane) {
  const half_t* p = tile + lane * 16;
  v8h lo = *(const v8h*)p;
  v8h hi = *(const v8h*)(p + 8);
  return make_v16(lo, hi);
}

// out[M,N] (16 x NB*16, row stride 64) = A(16xK) * W^T + bias
template<int NB, int KB, int AS>
__device__ __forceinline__ void mm_bias_store(float* outp, const half_t* a,
                                              const half_t* wpk, const float* bias, int lane) {
  const int N  = lane & 15;
  const int Mh = (lane >> 4) << 3;
#pragma unroll
  for (int nb = 0; nb < NB; ++nb) {
    const float bv = bias[nb * 16 + N];
    v8f acc = {bv, bv, bv, bv, bv, bv, bv, bv};
#pragma unroll
    for (int kb = 0; kb < KB; ++kb) {
      v16h A = load_a_frag(a, AS, kb, lane);
      v16h B = load_b_frag(wpk + (nb * KB + kb) * 512, lane);
      acc = __builtin_amdgcn_wmma_f32_16x16x32_f16(false, A, false, B, (short)0, acc, false, false);
    }
#pragma unroll
    for (int r = 0; r < 8; ++r)
      outp[(Mh + r) * 64 + nb * 16 + N] = acc[r];
  }
}

// Pack fp32 weight (N x K, row-major, used as x @ W^T) into f16 B-fragment order.
__device__ __forceinline__ void pack_weight(half_t* dst, const float* src, int N, int K, int tid) {
  const int KB = K >> 5;
  const int total = N * K;
  for (int p = tid; p < total; p += THREADS) {
    const int h = p & 15, l = (p >> 4) & 31, tile = p >> 9;
    const int kb = tile % KB, nb = tile / KB;
    const int n = nb * 16 + (l & 15);
    const int k = kb * 32 + ((l >> 4) << 3) + (h & 7) + ((h >> 3) << 4);
    dst[p] = (half_t)src[n * K + k];
  }
}

// dst = rmsnorm(x + y, nw); also writes f16 copy into a16 (A staging for next GEMM).
__device__ __forceinline__ void rmsnorm_add(float* dst, half_t* a16, const float* x,
                                            const float* y, const float* nw, int lane) {
  const int row = lane >> 1, off = (lane & 1) << 5;
  const float* px = x + row * 64 + off;
  const float* py = y + row * 64 + off;
  float ss = 0.f;
#pragma unroll
  for (int j = 0; j < 32; ++j) { float v = px[j] + py[j]; ss += v * v; }
  ss += __shfl_xor(ss, 1, 32);
  const float sc = rsqrtf(ss * (1.0f / 64.0f) + 1e-6f);
  float*  pd = dst + row * 64 + off;
  half_t* pa = a16 + row * 64 + off;
#pragma unroll
  for (int j = 0; j < 32; ++j) {
    float v = (px[j] + py[j]) * sc * nw[off + j];
    pd[j] = v;
    pa[j] = (half_t)v;
  }
}

// One Worker/Planner block on this wave's 16 rows. cbuf/a64 pre-filled with input c.
__device__ __forceinline__ void run_block(float* cbuf, float* obuf, half_t* a64, half_t* a256,
                                          const half_t* wset, const float* bset,
                                          float* dst, int lane) {
  // v = c @ Wv^T + bv
  mm_bias_store<4, 2, 64>(obuf, a64, wset + WV_H, bset + 0, lane);
  __syncthreads();
#pragma unroll
  for (int j = 0; j < 32; ++j) { int i = lane * 32 + j; a64[i] = (half_t)obuf[i]; }
  __syncthreads();
  // attn = v @ Ow^T + ob
  mm_bias_store<4, 2, 64>(obuf, a64, wset + OW_H, bset + 64, lane);
  __syncthreads();
  // h = rmsnorm(c + attn, n1) -> cbuf (+f16 into a64)
  rmsnorm_add(cbuf, a64, cbuf, obuf, bset + 448, lane);
  __syncthreads();
  // g = gelu(h @ F1^T + f1b) -> a256 (f16)
  {
    const int N = lane & 15, Mh = (lane >> 4) << 3;
#pragma unroll
    for (int nb = 0; nb < 16; ++nb) {
      const float bv = bset[128 + nb * 16 + N];
      v8f acc = {bv, bv, bv, bv, bv, bv, bv, bv};
#pragma unroll
      for (int kb = 0; kb < 2; ++kb) {
        v16h A = load_a_frag(a64, 64, kb, lane);
        v16h B = load_b_frag(wset + F1_H + (nb * 2 + kb) * 512, lane);
        acc = __builtin_amdgcn_wmma_f32_16x16x32_f16(false, A, false, B, (short)0, acc, false, false);
      }
#pragma unroll
      for (int r = 0; r < 8; ++r) {
        float x = acc[r];
        float g = 0.5f * x * (1.0f + erff(x * 0.70710678118654752f));
        a256[(Mh + r) * 256 + nb * 16 + N] = (half_t)g;
      }
    }
  }
  __syncthreads();
  // ffn = g @ F2^T + f2b
  mm_bias_store<4, 8, 256>(obuf, a256, wset + F2_H, bset + 384, lane);
  __syncthreads();
  // out = rmsnorm(h + ffn, n2)
  rmsnorm_add(dst, a64, cbuf, obuf, bset + 512, lane);
  __syncthreads();
}

__global__ void __launch_bounds__(THREADS) hrm4x4_fused(Params P) {
  extern __shared__ char smem[];
  half_t* wpk  = (half_t*)smem;
  float*  bias = (float*)(smem + BIAS_OFF);
  const int tid  = threadIdx.x;
  const int lane = tid & 31, wave = tid >> 5;
  char* wb = smem + WAVE_OFF + wave * WAVE_STRIDE;
  float*  hL   = (float*)(wb + HL_B);
  float*  hH   = (float*)(wb + HH_B);
  float*  xin  = (float*)(wb + XIN_B);
  float*  cbuf = (float*)(wb + CB_B);
  float*  obuf = (float*)(wb + OB_B);
  half_t* a64  = (half_t*)(wb + A64_B);
  half_t* a256 = (half_t*)(wb + A256_B);
  const int row0 = blockIdx.x * ROWS_PER_WG + wave * 16;

  // ---- init: x_in via WMMA (proj packed into planner-F2 slot as a temp) ----
  half_t* projpk = wpk + SET_H + F2_H;
  pack_weight(projpk, P.proj_w, 64, 256, tid);
  {
    const int M = lane & 15, c0 = (lane >> 4) * 8;
    const int* pz = P.puzzle + (size_t)(row0 + M) * 16;
    for (int c = c0; c < c0 + 8; ++c) {
      const int idx = pz[c];
      const float* e = P.emb + idx * 16;
#pragma unroll
      for (int j = 0; j < 16; ++j) a256[M * 256 + c * 16 + j] = (half_t)e[j];
    }
  }
  __syncthreads();
  mm_bias_store<4, 8, 256>(xin, a256, projpk, P.proj_b, lane);
  __syncthreads();

  // ---- biases / norm weights to LDS ----
  for (int i = tid; i < 64; i += THREADS) {
    bias[i]       = P.w_inb[128 + i]; bias[576 + i]       = P.p_inb[128 + i];
    bias[64 + i]  = P.w_ob[i];        bias[576 + 64 + i]  = P.p_ob[i];
    bias[384 + i] = P.w_f2b[i];       bias[576 + 384 + i] = P.p_f2b[i];
    bias[448 + i] = P.w_n1[i];        bias[576 + 448 + i] = P.p_n1[i];
    bias[512 + i] = P.w_n2[i];        bias[576 + 512 + i] = P.p_n2[i];
  }
  for (int i = tid; i < 256; i += THREADS) {
    bias[128 + i] = P.w_f1b[i];       bias[576 + 128 + i] = P.p_f1b[i];
  }
  // ---- pack both weight sets (overwrites projpk; safe after xin sync) ----
  pack_weight(wpk + WV_H,         P.w_inw + 128 * 64, 64, 64, tid);
  pack_weight(wpk + OW_H,         P.w_ow,             64, 64, tid);
  pack_weight(wpk + F1_H,         P.w_f1w,            256, 64, tid);
  pack_weight(wpk + F2_H,         P.w_f2w,            64, 256, tid);
  pack_weight(wpk + SET_H + WV_H, P.p_inw + 128 * 64, 64, 64, tid);
  pack_weight(wpk + SET_H + OW_H, P.p_ow,             64, 64, tid);
  pack_weight(wpk + SET_H + F1_H, P.p_f1w,            256, 64, tid);
  pack_weight(wpk + SET_H + F2_H, P.p_f2w,            64, 256, tid);
  // ---- init states ----
#pragma unroll
  for (int j = 0; j < 32; ++j) {
    int i = lane * 32 + j;
    hL[i] = P.hL0[i & 63];
    hH[i] = P.hH0[i & 63];
  }
  __syncthreads();

  // ---- recurrence: 5 cycles x (10 worker steps + 1 planner step) ----
#pragma unroll 1
  for (int cyc = 0; cyc < 5; ++cyc) {
#pragma unroll 1
    for (int st = 0; st < 10; ++st) {
#pragma unroll
      for (int j = 0; j < 32; ++j) {
        int i = lane * 32 + j;
        float x = hL[i] + hH[i] + xin[i];
        cbuf[i] = x; a64[i] = (half_t)x;
      }
      __syncthreads();
      run_block(cbuf, obuf, a64, a256, wpk, bias, hL, lane);
    }
#pragma unroll
    for (int j = 0; j < 32; ++j) {
      int i = lane * 32 + j;
      float x = hH[i] + hL[i];
      cbuf[i] = x; a64[i] = (half_t)x;
    }
    __syncthreads();
    run_block(cbuf, obuf, a64, a256, wpk + SET_H, bias + 576, hH, lane);
  }

  // ---- logits: 16 rows x (16 cell + 4 digit) = 320 dots per wave ----
  for (int t = lane; t < 320; t += 32) {
    const int r = t / 20, o = t % 20;
    const float* hr = hH + r * 64;
    const int grow = row0 + r;
    const float* w; float b; float* dp;
    if (o < 16) {
      w = P.cell_w + o * 64; b = P.cell_b[o];
      dp = P.out + (size_t)grow * 16 + o;
    } else {
      const int d = o - 16;
      w = P.dig_w + d * 64; b = P.dig_b[d];
      dp = P.out + (size_t)P.nrows * 16 + (size_t)grow * 4 + d;
    }
    float s = b;
#pragma unroll
    for (int k = 0; k < 64; ++k) s += hr[k] * w[k];
    *dp = s;
  }
}

extern "C" void kernel_launch(void* const* d_in, const int* in_sizes, int n_in,
                              void* d_out, int out_size, void* d_ws, size_t ws_size,
                              hipStream_t stream) {
  (void)n_in; (void)out_size; (void)d_ws; (void)ws_size;
  Params P;
  P.puzzle = (const int*)d_in[0];
  P.emb    = (const float*)d_in[1];
  P.proj_w = (const float*)d_in[2];
  P.proj_b = (const float*)d_in[3];
  P.hL0    = (const float*)d_in[4];
  P.hH0    = (const float*)d_in[5];
  P.cell_w = (const float*)d_in[6];
  P.cell_b = (const float*)d_in[7];
  P.dig_w  = (const float*)d_in[8];
  P.dig_b  = (const float*)d_in[9];
  P.w_inw  = (const float*)d_in[10]; P.w_inb = (const float*)d_in[11];
  P.w_ow   = (const float*)d_in[12]; P.w_ob  = (const float*)d_in[13];
  P.w_f1w  = (const float*)d_in[14]; P.w_f1b = (const float*)d_in[15];
  P.w_f2w  = (const float*)d_in[16]; P.w_f2b = (const float*)d_in[17];
  P.w_n1   = (const float*)d_in[18]; P.w_n2  = (const float*)d_in[19];
  P.p_inw  = (const float*)d_in[20]; P.p_inb = (const float*)d_in[21];
  P.p_ow   = (const float*)d_in[22]; P.p_ob  = (const float*)d_in[23];
  P.p_f1w  = (const float*)d_in[24]; P.p_f1b = (const float*)d_in[25];
  P.p_f2w  = (const float*)d_in[26]; P.p_f2b = (const float*)d_in[27];
  P.p_n1   = (const float*)d_in[28]; P.p_n2  = (const float*)d_in[29];
  P.out    = (float*)d_out;
  P.nrows  = in_sizes[0] / 16;
  const int grid = P.nrows / ROWS_PER_WG;
  hipLaunchKernelGGL(hrm4x4_fused, dim3(grid), dim3(THREADS), LDS_BYTES, stream, P);
}
